// ResidualLSTMLayer_88587995447885
// MI455X (gfx1250) — matrix operand order repaired and checked by reading, not compiled, not run on hardware
//
#include <hip/hip_runtime.h>
#include <hip/hip_bf16.h>
#include <stdint.h>

typedef __attribute__((ext_vector_type(16))) __bf16 v16bf;
typedef __attribute__((ext_vector_type(8)))  float  v8f;

#define T_STEPS 512
#define NWG     32
#define THREADS 256
#define WAVES_PER_WG 8

// ---------------------------------------------------------------------------
// Helpers
// ---------------------------------------------------------------------------
__device__ __forceinline__ uint32_t pack_bf16x2(float a, float b) {
  uint32_t ua = __float_as_uint(a);
  uint32_t ub = __float_as_uint(b);
  ua = (ua + 0x7FFFu + ((ua >> 16) & 1u)) >> 16;  // RNE f32->bf16
  ub = (ub + 0x7FFFu + ((ub >> 16) & 1u)) >> 16;
  return ua | (ub << 16);
}
__device__ __forceinline__ uint16_t to_bf16(float a) {
  uint32_t u = __float_as_uint(a);
  return (uint16_t)((u + 0x7FFFu + ((u >> 16) & 1u)) >> 16);
}
__device__ __forceinline__ float fast_sigmoid(float x) {
  return 1.0f / (1.0f + __expf(-x));
}
__device__ __forceinline__ float fast_tanh(float x) {
  float e = __expf(-2.0f * x);
  return (1.0f - e) / (1.0f + e);
}
// A-fragment K offset within a 32-K block for (vgpr v, lane-group g, half h):
//   k32 = (v<4 ? 2v : 2(v-4)+16) + 8g + h   (WMMA 16-bit A 16x32 layout)
__device__ __forceinline__ unsigned afrag_kbase(unsigned v, unsigned lgrp) {
  return (v < 4u ? 2u * v : 2u * (v - 4u) + 16u) + 8u * lgrp;
}

// ---------------------------------------------------------------------------
// Weight repack: combined B [K=2048][N=4096] -> bf16 WMMA B-fragments.
// Fragment (kb 0..63, gate g 0..3, jt 0..63): 32 lanes x 8 dwords, per-lane
// contiguous. B 32x16 16-bit layout: k32 = (v<4 ? 2v : 2(v-4)+16) + 8*(lane/16) + half.
// ---------------------------------------------------------------------------
__global__ void prep_weights(const float* __restrict__ Wih,
                             const float* __restrict__ Whh,
                             uint32_t* __restrict__ wfrag) {
  const uint32_t total = 64u * 4u * 64u * 256u; // 4,194,304 dwords = 16 MB
  for (uint32_t idx = blockIdx.x * blockDim.x + threadIdx.x; idx < total;
       idx += gridDim.x * blockDim.x) {
    uint32_t within = idx & 255u;
    uint32_t frag   = idx >> 8;
    uint32_t lane = within >> 3, v = within & 7u;
    uint32_t jt = frag & 63u;
    uint32_t g  = (frag >> 6) & 3u;
    uint32_t kb = frag >> 8;                       // 0..63 over combined K
    uint32_t nn = lane & 15u, kg = lane >> 4;
    uint32_t k0  = kb * 32u + afrag_kbase(v, kg);
    uint32_t row = g * 1024u + jt * 16u + nn;
    float f0, f1;
    if (k0 < 1024u) {
      f0 = Wih[(size_t)row * 1024u + k0];
      f1 = Wih[(size_t)row * 1024u + k0 + 1u];
    } else {
      f0 = Whh[(size_t)row * 1024u + (k0 - 1024u)];
      f1 = Whh[(size_t)row * 1024u + (k0 - 1024u) + 1u];
    }
    wfrag[idx] = pack_bf16x2(f0, f1);
  }
}

// ---------------------------------------------------------------------------
// Pre-fragment the ENTIRE input into WMMA A layout (parallel over all T).
// xfrag[t][kb 0..31][mt 0..3][lane][v] dwords, 32768 dwords (128KB) per step.
// ---------------------------------------------------------------------------
__global__ void prep_x(const float* __restrict__ x, uint32_t* __restrict__ xfrag) {
  const uint32_t total = (uint32_t)T_STEPS * 32768u; // 16,777,216 dwords = 64 MB
  for (uint32_t idx = blockIdx.x * blockDim.x + threadIdx.x; idx < total;
       idx += gridDim.x * blockDim.x) {
    uint32_t t    = idx >> 15;
    uint32_t rem  = idx & 32767u;
    uint32_t kb   = rem >> 10;
    uint32_t mtile = (rem >> 8) & 3u;
    uint32_t lane = (rem >> 3) & 31u;
    uint32_t v    = rem & 7u;
    uint32_t k = kb * 32u + afrag_kbase(v, lane >> 4);
    uint32_t m = mtile * 16u + (lane & 15u);
    size_t sb = (size_t)t * 65536u + (size_t)m * 1024u + k;
    xfrag[idx] = pack_bf16x2(x[sb], x[sb + 1]);
  }
}

// hx -> A-fragment layout into hfrag ping buffer 0 (read at t=0).
__global__ void prep_h(const float* __restrict__ hx, uint32_t* __restrict__ hfrag0) {
  uint32_t idx = blockIdx.x * blockDim.x + threadIdx.x;
  if (idx >= 32768u) return;
  uint32_t kb   = idx >> 10;
  uint32_t mtile = (idx >> 8) & 3u;
  uint32_t lane = (idx >> 3) & 31u;
  uint32_t v    = idx & 7u;
  uint32_t k = kb * 32u + afrag_kbase(v, lane >> 4);
  uint32_t m = mtile * 16u + (lane & 15u);
  hfrag0[idx] = pack_bf16x2(hx[m * 1024u + k], hx[m * 1024u + k + 1u]);
}

__global__ void prep_bias(const float* __restrict__ bih,
                          const float* __restrict__ bhh,
                          float* __restrict__ bsum) {
  int i = blockIdx.x * blockDim.x + threadIdx.x;
  if (i < 4096) bsum[i] = bih[i] + bhh[i];
}

// ---------------------------------------------------------------------------
// Persistent recurrent kernel. 32 WGs x 8 waves = 256 waves.
// Wave w: mt = w&3 (batch tile), jt = w>>2 (hidden tile) -> the 4 waves that
// share each B fragment are in the SAME workgroup (WGP$ reuse).
// Per step: 64 kblocks x 4 gates = 256 WMMAs/wave, all operands preswizzled in
// global memory (no LDS staging, no intra-step barriers), fused LSTM epilogue
// that scatters h directly in bf16 A-fragment layout for the next step.
// ---------------------------------------------------------------------------
__launch_bounds__(THREADS, 1)
__global__ void lstm_persistent(const float* __restrict__ x,
                                const float* __restrict__ cx,
                                const uint32_t* __restrict__ wfrag,
                                const uint32_t* __restrict__ xfrag,
                                const float* __restrict__ bsum,
                                uint32_t* __restrict__ hfrag0,
                                uint32_t* __restrict__ hfrag1,
                                float* __restrict__ cbuf,
                                unsigned* __restrict__ counter,
                                float* __restrict__ out) {
  const unsigned tid  = threadIdx.x;
  const unsigned lane = tid & 31u;
  const unsigned wave = tid >> 5;
  const unsigned w    = blockIdx.x * WAVES_PER_WG + wave; // 0..255
  const unsigned mt   = w & 3u;    // batch-row tile
  const unsigned jt   = w >> 2;    // hidden-column tile 0..63
  const unsigned nn   = lane & 15u;
  const unsigned ncol = jt * 16u + nn;

  float bias_g[4];
#pragma unroll
  for (int g = 0; g < 4; ++g) bias_g[g] = bsum[g * 1024 + ncol];

  // constant pieces of the h-fragment scatter address for this lane
  const unsigned kbh   = jt >> 1;
  const unsigned vh    = ((nn & 7u) >> 1) + 4u * (jt & 1u);
  const unsigned halfh = nn & 1u;
  const unsigned lgrp  = (nn >= 8u) ? 16u : 0u;

  const unsigned aoff = mt * 256u + lane * 8u;   // + kb*1024 per kblock
  const unsigned boff = jt * 256u + lane * 8u;   // + (kb*4+g)*16384 per frag

  for (int t = 0; t < T_STEPS; ++t) {
    // make this step's h-fragments (written by other WGPs) visible: L0 inv
    __builtin_amdgcn_fence(__ATOMIC_ACQUIRE, "agent");

    v8f acc[4];
#pragma unroll
    for (int g = 0; g < 4; ++g)
#pragma unroll
      for (int r = 0; r < 8; ++r) acc[g][r] = bias_g[g];

    const uint32_t* xf = xfrag + (size_t)t * 32768u;
    const uint32_t* hf = (t & 1) ? hfrag1 : hfrag0;

    // ---- K = 0..1023 : input projection (preswizzled xfrag) ----
#pragma unroll 8
    for (unsigned kb = 0; kb < 32u; ++kb) {
      const v16bf a = *reinterpret_cast<const v16bf*>(xf + kb * 1024u + aoff);
#pragma unroll
      for (unsigned g = 0; g < 4u; ++g) {
        const v16bf b = *reinterpret_cast<const v16bf*>(
            wfrag + (kb * 4u + g) * 16384u + boff);
        acc[g] = __builtin_amdgcn_wmma_f32_16x16x32_bf16(
            false, a, false, b, (short)0, acc[g], false, false);
      }
    }
    // ---- K = 1024..2047 : recurrent projection (hfrag ping-pong) ----
#pragma unroll 8
    for (unsigned kb = 0; kb < 32u; ++kb) {
      const v16bf a = *reinterpret_cast<const v16bf*>(hf + kb * 1024u + aoff);
#pragma unroll
      for (unsigned g = 0; g < 4u; ++g) {
        const v16bf b = *reinterpret_cast<const v16bf*>(
            wfrag + ((kb + 32u) * 4u + g) * 16384u + boff);
        acc[g] = __builtin_amdgcn_wmma_f32_16x16x32_bf16(
            false, a, false, b, (short)0, acc[g], false, false);
      }
    }

    // ---- fused LSTM epilogue; scatter h as next step's A fragments ----
    uint16_t* hw16 = (uint16_t*)((t & 1) ? hfrag0 : hfrag1);
#pragma unroll
    for (int r = 0; r < 8; ++r) {
      unsigned mr   = (unsigned)r + ((lane >> 4) << 3);
      unsigned row  = mt * 16u + mr;
      unsigned cidx = row * 1024u + ncol;
      float cold = (t == 0) ? cx[cidx] : cbuf[cidx];
      float ig = fast_sigmoid(acc[0][r]);
      float fg = fast_sigmoid(acc[1][r]);
      float gt = fast_tanh   (acc[2][r]);
      float og = fast_sigmoid(acc[3][r]);
      float cn = fg * cold + ig * gt;
      float hn = og * fast_tanh(cn);
      cbuf[cidx] = cn;                         // wave-private, in place
      // h -> bf16 A-fragment scatter (element (m=row, k=ncol) of next A)
      unsigned didx = ((kbh * 4u + mt) * 32u + mr + lgrp) * 16u + vh * 2u + halfh;
      hw16[didx] = to_bf16(hn);
      size_t oidx = (size_t)t * 65536u + cidx;
      out[oidx] = hn + x[oidx];                // residual
      if (t == T_STEPS - 1) {
        out[(size_t)T_STEPS * 65536u + cidx]          = hn;  // final h
        out[(size_t)T_STEPS * 65536u + 65536u + cidx] = cn;  // final c
      }
    }

    // ---- device-wide step barrier (release h writes, then rendezvous) ----
    __threadfence();
    __syncthreads();
    if (tid == 0) {
      unsigned target = (unsigned)(t + 1) * NWG;
      __hip_atomic_fetch_add(counter, 1u, __ATOMIC_ACQ_REL,
                             __HIP_MEMORY_SCOPE_AGENT);
      while (__hip_atomic_load(counter, __ATOMIC_ACQUIRE,
                               __HIP_MEMORY_SCOPE_AGENT) < target) {
        __builtin_amdgcn_s_sleep(8);
      }
    }
    __syncthreads();
  }
}

// ---------------------------------------------------------------------------
// Launch
// ---------------------------------------------------------------------------
extern "C" void kernel_launch(void* const* d_in, const int* in_sizes, int n_in,
                              void* d_out, int out_size, void* d_ws, size_t ws_size,
                              hipStream_t stream) {
  (void)in_sizes; (void)n_in; (void)out_size; (void)ws_size;
  const float* input = (const float*)d_in[0];
  const float* hx    = (const float*)d_in[1];
  const float* cx    = (const float*)d_in[2];
  const float* Wih   = (const float*)d_in[3];
  const float* Whh   = (const float*)d_in[4];
  const float* bih   = (const float*)d_in[5];
  const float* bhh   = (const float*)d_in[6];
  float* out = (float*)d_out;

  char* ws = (char*)d_ws;
  uint32_t* wfrag   = (uint32_t*)ws;                          // 16 MB
  uint32_t* xfrag   = (uint32_t*)(ws + (16u << 20));          // 64 MB
  float*    bsum    = (float*)(ws + (80u << 20));             // 16 KB
  uint32_t* hfrag0  = (uint32_t*)(ws + (80u << 20) + (1u << 16)); // 128 KB
  uint32_t* hfrag1  = hfrag0 + 32768;                         // 128 KB
  float*    cbuf    = (float*)(hfrag1 + 32768);               // 256 KB
  unsigned* counter = (unsigned*)(cbuf + 65536);

  hipMemsetAsync(counter, 0, 64, stream);
  prep_weights<<<4096, 256, 0, stream>>>(Wih, Whh, wfrag);
  prep_x<<<8192, 256, 0, stream>>>(input, xfrag);
  prep_h<<<128, 256, 0, stream>>>(hx, hfrag0);
  prep_bias<<<16, 256, 0, stream>>>(bih, bhh, bsum);
  lstm_persistent<<<NWG, THREADS, 0, stream>>>(input, cx, wfrag, xfrag, bsum,
                                               hfrag0, hfrag1, cbuf, counter, out);
}
